// ASH_DGT_31791347925004
// MI455X (gfx1250) — 8-bit matrix operand order repaired and checked by reading, compiled, not run on hardware
//
#include <hip/hip_runtime.h>
#include <math.h>

#define Nn     2048
#define Dd     512
#define Ms     512
#define NPAIRS (Nn * (Nn - 1) / 2)
#define SORTN  (1 << 21)
#define TARGET (Nn - Ms)
#define LBN    2048

typedef __attribute__((ext_vector_type(8))) int v8i;
typedef unsigned long long u64;

__global__ __launch_bounds__(256) void k_binarize(const float* __restrict__ A,
                                                  unsigned char* __restrict__ A8) {
    size_t t = (size_t)blockIdx.x * 256 + threadIdx.x;
    float4 f = reinterpret_cast<const float4*>(A)[t];
    uchar4 o = make_uchar4(f.x > 0.0f, f.y > 0.0f, f.z > 0.0f, f.w > 0.0f);
    reinterpret_cast<uchar4*>(A8)[t] = o;
}

__global__ __launch_bounds__(256) void k_deg(const unsigned char* __restrict__ A8,
                                             int* __restrict__ deg) {
    __shared__ int sh;
    if (threadIdx.x == 0) sh = 0;
    __syncthreads();
    const unsigned char* r = A8 + (size_t)blockIdx.x * Nn;
    int s = 0;
    for (int k = threadIdx.x; k < Nn; k += 256) s += r[k];
    atomicAdd(&sh, s);
    __syncthreads();
    if (threadIdx.x == 0) deg[blockIdx.x] = sh;
}

__global__ __launch_bounds__(256) void k_initkeys(u64* __restrict__ keys) {
    int t = blockIdx.x * 256 + threadIdx.x;
    keys[t] = ~0ull;
}

__global__ __launch_bounds__(128)
void k_jaccard_wmma(const unsigned char* __restrict__ A8,
                    const int* __restrict__ deg,
                    u64* __restrict__ keys) {
    const int wave = threadIdx.x >> 5;
    const int lane = threadIdx.x & 31;
    const int tI = blockIdx.y;
    const int tJ = blockIdx.x * 4 + wave;
    if (tJ < tI) return;
    const int lo = lane & 15, hi = lane >> 4;
    const int rowBase = tI * 16, colBase = tJ * 16;

    const unsigned char* aRow = A8 + (size_t)(rowBase + lo) * Nn;
    const unsigned char* bRow = A8 + (size_t)(colBase + lo) * Nn;

    v8i acc = {0, 0, 0, 0, 0, 0, 0, 0};
    for (int k0 = 0; k0 < Nn; k0 += 64) {
        v8i a, b;
#pragma unroll
        for (int v = 0; v < 8; ++v) {
            const int ka = ((v >> 1) << 4) + ((v & 1) << 2) + (hi << 3);
            a[v] = *reinterpret_cast<const int*>(aRow + k0 + ka);
            const int kb = ((v >> 1) << 4) + ((v & 1) << 2) + (hi << 3);
            b[v] = *reinterpret_cast<const int*>(bRow + k0 + kb);
        }
        __builtin_prefetch(aRow + k0 + 64, 0, 1);
        __builtin_prefetch(bRow + k0 + 64, 0, 1);
        acc = __builtin_amdgcn_wmma_i32_16x16x64_iu8(false, a, false, b, acc,
                                                     false, false);
    }

    const int j  = colBase + lo;
    const int dj = deg[j];
#pragma unroll
    for (int r = 0; r < 8; ++r) {
        const int i = rowBase + r + hi * 8;
        if (i < j) {
            const int inter = acc[r];
            const int u = deg[i] + dj - inter;
            const float jac = (u == 0) ? 0.0f : (float)inter / (float)u;
            const unsigned sb = __float_as_uint(jac);
            const u64 key = ((u64)(~sb) << 32) | (unsigned)((i << 16) | j);
            const long long p =
                (long long)i * (Nn - 1) - ((long long)i * (i - 1)) / 2 + (j - i - 1);
            keys[p] = key;
        }
    }
}

__global__ __launch_bounds__(256) void k_bitonic(u64* __restrict__ keys, int jj, int kk) {
    int i = blockIdx.x * 256 + threadIdx.x;
    int ixj = i ^ jj;
    if (ixj > i) {
        u64 a = keys[i], b = keys[ixj];
        bool up = ((i & kk) == 0);
        if ((a > b) == up) { keys[i] = b; keys[ixj] = a; }
    }
}

__global__ __launch_bounds__(256) void k_bitonic_local_first(u64* __restrict__ keys) {
    __shared__ u64 sh[LBN];
    const int base = blockIdx.x * LBN;
    for (int t = threadIdx.x; t < LBN; t += 256) sh[t] = keys[base + t];
    __syncthreads();
    for (int kk = 2; kk <= LBN; kk <<= 1) {
        for (int jj = kk >> 1; jj > 0; jj >>= 1) {
#pragma unroll
            for (int q = 0; q < LBN / 512; ++q) {
                const int t = threadIdx.x + q * 256;
                const int i = ((t & ~(jj - 1)) << 1) | (t & (jj - 1));
                const bool up = (((base + i) & kk) == 0);
                u64 a = sh[i], b = sh[i + jj];
                if ((a > b) == up) { sh[i] = b; sh[i + jj] = a; }
            }
            __syncthreads();
        }
    }
    for (int t = threadIdx.x; t < LBN; t += 256) keys[base + t] = sh[t];
}

__global__ __launch_bounds__(256) void k_bitonic_local(u64* __restrict__ keys, int kk) {
    __shared__ u64 sh[LBN];
    const int base = blockIdx.x * LBN;
    for (int t = threadIdx.x; t < LBN; t += 256) sh[t] = keys[base + t];
    __syncthreads();
    const bool up = ((base & kk) == 0);
    for (int jj = LBN / 2; jj > 0; jj >>= 1) {
#pragma unroll
        for (int q = 0; q < LBN / 512; ++q) {
            const int t = threadIdx.x + q * 256;
            const int i = ((t & ~(jj - 1)) << 1) | (t & (jj - 1));
            u64 a = sh[i], b = sh[i + jj];
            if ((a > b) == up) { sh[i] = b; sh[i + jj] = a; }
        }
        __syncthreads();
    }
    for (int t = threadIdx.x; t < LBN; t += 256) keys[base + t] = sh[t];
}

__global__ __launch_bounds__(32)
void k_uf_label(const u64* __restrict__ keys, int* __restrict__ labels,
                float* __restrict__ invs) {
    __shared__ int par[Nn];
    __shared__ int labmap[Nn];
    __shared__ int lab[Nn];
    __shared__ int szs[Ms];
    __shared__ u64 buf[256];
    __shared__ int done;
    const int lane = threadIdx.x;
    for (int t = lane; t < Nn; t += 32) { par[t] = t; labmap[t] = -1; }
    if (lane == 0) done = 0;
    __syncthreads();

    int merged = 0;
    for (int s0 = 0; s0 < NPAIRS; s0 += 256) {
#pragma unroll
        for (int q = 0; q < 8; ++q) {
            const int idx = s0 + q * 32 + lane;
            buf[q * 32 + lane] = (idx < NPAIRS) ? keys[idx] : ~0ull;
        }
        __syncthreads();
        if (lane == 0) {
            for (int q = 0; q < 256; ++q) {
                const unsigned pj = (unsigned)buf[q];
                const int i = pj >> 16, j = pj & 0xFFFF;
                if (i >= Nn || j >= Nn) continue;
                int ri = i; while (par[ri] != ri) ri = par[ri];
                int rj = j; while (par[rj] != rj) rj = par[rj];
                if (ri != rj) {
                    par[rj] = ri;
                    if (++merged >= TARGET) { done = 1; break; }
                }
            }
        }
        __syncthreads();
        if (done) break;
    }

    if (lane == 0) {
        for (int c = 0; c < Ms; ++c) szs[c] = 0;
        int cnt = 0;
        for (int i = 0; i < Nn; ++i) {
            int r = i; while (par[r] != r) r = par[r];
            int l = labmap[r];
            if (l < 0) { l = (cnt < Ms) ? cnt : Ms - 1; labmap[r] = l; ++cnt; }
            lab[i] = l;
            szs[l] += 1;
        }
    }
    __syncthreads();
    for (int t = lane; t < Nn; t += 32) labels[t] = lab[t];
    for (int t = lane; t < Ms; t += 32)
        invs[t] = 1.0f / sqrtf((float)szs[t] + 1e-10f);
}

__global__ __launch_bounds__(256) void k_P(const int* __restrict__ labels,
                                           const float* __restrict__ invs,
                                           float* __restrict__ P) {
    int idx = blockIdx.x * 256 + threadIdx.x;
    int i = idx >> 9, c = idx & (Ms - 1);
    P[idx] = (labels[i] == c) ? invs[c] : 0.0f;
}

__global__ __launch_bounds__(128) void k_Xc(const float* __restrict__ X,
                                            const int* __restrict__ labels,
                                            const float* __restrict__ invs,
                                            float* __restrict__ Xc) {
    __shared__ int lab[Nn];
    const int c = blockIdx.x;
    for (int t = threadIdx.x; t < Nn; t += 128) lab[t] = labels[t];
    __syncthreads();
    const int d0 = threadIdx.x;
    float s0 = 0, s1 = 0, s2 = 0, s3 = 0;
    for (int i = 0; i < Nn; ++i)
        if (lab[i] == c) {
            const float* xr = X + (size_t)i * Dd;
            s0 += xr[d0]; s1 += xr[d0 + 128]; s2 += xr[d0 + 256]; s3 += xr[d0 + 384];
        }
    const float w = invs[c];
    float* o = Xc + (size_t)c * Dd;
    o[d0] = s0 * w; o[d0 + 128] = s1 * w; o[d0 + 256] = s2 * w; o[d0 + 384] = s3 * w;
}

__global__ __launch_bounds__(256) void k_M1(const float* __restrict__ A,
                                            const int* __restrict__ labels,
                                            const float* __restrict__ invs,
                                            float* __restrict__ M1) {
    __shared__ int lab[Nn];
    const int c = blockIdx.x;
    for (int t = threadIdx.x; t < Nn; t += 256) lab[t] = labels[t];
    __syncthreads();
    float s[8] = {0, 0, 0, 0, 0, 0, 0, 0};
    for (int i = 0; i < Nn; ++i)
        if (lab[i] == c) {
            const float* ar = A + (size_t)i * Nn;
#pragma unroll
            for (int t = 0; t < 8; ++t) s[t] += ar[threadIdx.x + t * 256];
        }
    const float w = invs[c];
#pragma unroll
    for (int t = 0; t < 8; ++t)
        M1[(size_t)c * Nn + threadIdx.x + t * 256] = s[t] * w;
}

__global__ __launch_bounds__(256) void k_Ac(const float* __restrict__ M1,
                                            const int* __restrict__ labels,
                                            const float* __restrict__ invs,
                                            float* __restrict__ Ac) {
    __shared__ int lab[Nn];
    __shared__ float row[Nn];
    const int c = blockIdx.x;
    for (int t = threadIdx.x; t < Nn; t += 256) {
        lab[t] = labels[t];
        row[t] = M1[(size_t)c * Nn + t];
    }
    __syncthreads();
    const int e0 = threadIdx.x, e1 = threadIdx.x + 256;
    float s0 = 0, s1 = 0;
    for (int j = 0; j < Nn; ++j) {
        const int l = lab[j];
        const float m = row[j];
        s0 += (l == e0) ? m : 0.0f;
        s1 += (l == e1) ? m : 0.0f;
    }
    Ac[(size_t)c * Ms + e0] = s0 * invs[e0];
    Ac[(size_t)c * Ms + e1] = s1 * invs[e1];
}

extern "C" void kernel_launch(void* const* d_in, const int* in_sizes, int n_in,
                              void* d_out, int out_size, void* d_ws, size_t ws_size,
                              hipStream_t stream) {
    (void)in_sizes; (void)n_in; (void)out_size; (void)ws_size;
    const float* X = (const float*)d_in[0];
    const float* A = (const float*)d_in[1];

    float* Xc = (float*)d_out;
    float* Ac = Xc + (size_t)Ms * Dd;
    float* P  = Ac + (size_t)Ms * Ms;

    char* wsb = (char*)d_ws;
    unsigned char* A8 = (unsigned char*)wsb;
    u64*  keys = (u64*)(wsb + (4u << 20));
    float* M1  = (float*)keys;
    int*   labels = (int*)(wsb + (20u << 20));
    float* invs   = (float*)(labels + Nn);

    k_binarize<<<(Nn * Nn / 4) / 256, 256, 0, stream>>>(A, A8);
    k_deg<<<Nn, 256, 0, stream>>>(A8, labels);
    int* deg = labels;
    k_initkeys<<<SORTN / 256, 256, 0, stream>>>(keys);
    k_jaccard_wmma<<<dim3(Nn / 16 / 4, Nn / 16), 128, 0, stream>>>(A8, deg, keys);

    k_bitonic_local_first<<<SORTN / LBN, 256, 0, stream>>>(keys);
    for (int kk = 2 * LBN; kk <= SORTN; kk <<= 1) {
        for (int jj = kk >> 1; jj >= LBN; jj >>= 1)
            k_bitonic<<<SORTN / 256, 256, 0, stream>>>(keys, jj, kk);
        k_bitonic_local<<<SORTN / LBN, 256, 0, stream>>>(keys, kk);
    }

    k_uf_label<<<1, 32, 0, stream>>>(keys, labels, invs);

    k_P<<<(Nn * Ms) / 256, 256, 0, stream>>>(labels, invs, P);
    k_Xc<<<Ms, 128, 0, stream>>>(X, labels, invs, Xc);
    k_M1<<<Ms, 256, 0, stream>>>(A, labels, invs, M1);
    k_Ac<<<Ms, 256, 0, stream>>>(M1, labels, invs, Ac);
}
